// MultiHeadAttention_80401787781293
// MI455X (gfx1250) — matrix operand order repaired and checked by reading, not compiled, not run on hardware
//
#include <hip/hip_runtime.h>
#include <hip/hip_bf16.h>

// ---------------------------------------------------------------------------
// MI455X (gfx1250) multi-head attention, bf16 WMMA + TDM async tile staging.
//   B=4, N=2048, DIM=768, H=12, HD=64.  ~90 GFLOP / ~35MB -> compute bound ->
//   all matmuls on v_wmma_f32_16x16x32_bf16; GEMM tiles staged by the
//   Tensor Data Mover (tensor_load_to_lds, TENSORcnt), double buffered.
// ---------------------------------------------------------------------------

typedef __bf16 bf16_t;
typedef __attribute__((ext_vector_type(16))) __bf16 v16bf;
typedef __attribute__((ext_vector_type(8)))  __bf16 v8bf;
typedef __attribute__((ext_vector_type(8)))  float  v8f;
typedef __attribute__((ext_vector_type(4)))  unsigned int u32x4;
typedef __attribute__((ext_vector_type(8)))  int i32x8;
typedef __attribute__((ext_vector_type(4)))  int i32x4;

#define BATCH 4
#define NTOK  2048
#define DIM   768
#define NH    12
#define HD    64
#define QKVN  (3 * DIM)      // 2304
#define SCALE 0.125f         // 64^-0.5
#define LSTR  40             // bf16 LDS row stride: 80B, 16B-aligned, no bank repeat
#define SSTR  36             // f32 softmax staging stride: 144B, 16B-aligned

// ---- WMMA wrapper ---------------------------------------------------------
__device__ __forceinline__ v8f wmma_bf16(v16bf a, v16bf b, v8f c) {
    return __builtin_amdgcn_wmma_f32_16x16x32_bf16(
        false, a, false, b, (short)0, c, false, false);
}

// ---- fragment loaders (ISA 7.12.2 layouts) --------------------------------
// A (16x32, MxK): lane row M=L%16; K runs [hi*8,hi*8+8) u [hi*8+16,hi*8+24).
__device__ __forceinline__ v16bf load_frag_a(const bf16_t* base, int stride,
                                             int row, int hi) {
    union { v8bf h[2]; v16bf v; } u;
    const bf16_t* p = base + (size_t)row * stride + hi * 8;
    u.h[0] = *(const v8bf*)(p);
    u.h[1] = *(const v8bf*)(p + 16);
    return u.v;
}
// B (32x16, KxN from [n][k] storage): lane col N=L%16; K runs [hi*8,hi*8+8) u [hi*8+16,hi*8+24).
__device__ __forceinline__ v16bf load_frag_b(const bf16_t* base, int stride,
                                             int row_n, int hi) {
    union { v8bf h[2]; v16bf v; } u;
    const bf16_t* p = base + (size_t)row_n * stride + hi * 8;
    u.h[0] = *(const v8bf*)(p);
    u.h[1] = *(const v8bf*)(p + 16);
    return u.v;
}

// ---- Tensor Data Mover: 2D tile (32 x 128 rows, 2B elems) -> LDS ----------
// D# per CDNA5 ISA.  pad_interval=3 (16 dwords = 64B =
// one 32-elem row), pad_amount=3 (4 dwords = 16B) -> LDS pitch 80B == LSTR.
__device__ __forceinline__ uint32_t lds_off(const void* p) {
    return (uint32_t)(uintptr_t)p;   // flat LDS addr low 32 bits = LDS offset
}
__device__ __forceinline__ void tdm_load_tile(uint32_t lds_addr,
                                              const bf16_t* gptr,
                                              uint32_t tensor_d0,
                                              uint32_t stride_elems) {
    uint64_t ga = (uint64_t)(uintptr_t)gptr;
    u32x4 g0;
    g0[0] = 1u;                                   // count=1, user descriptor
    g0[1] = lds_addr;                             // lds_addr
    g0[2] = (uint32_t)ga;                         // global_addr[31:0]
    g0[3] = (uint32_t)(ga >> 32) | (2u << 30);    // global_addr[56:32] | type=2
    i32x8 g1;
    g1[0] = (int)((1u << 16) | (1u << 20) | (3u << 22) | (3u << 25));
             // data_size=2B | pad_enable | pad_interval=16dw | pad_amount=4dw
    g1[1] = (int)((tensor_d0 & 0xFFFFu) << 16);   // tensor_dim0[15:0]
    g1[2] = (int)((tensor_d0 >> 16) | (128u << 16)); // dim0 hi | tensor_dim1=128
    g1[3] = (int)(32u << 16);                     // tile_dim0 = 32
    g1[4] = 128;                                  // tile_dim1 = 128
    g1[5] = (int)stride_elems;                    // tensor_dim0_stride lo32
    g1[6] = 0;
    g1[7] = 0;
    i32x4 z4 = {0, 0, 0, 0};
#if defined(__clang_major__) && (__clang_major__ >= 23)
    i32x8 z8 = {0, 0, 0, 0, 0, 0, 0, 0};
    __builtin_amdgcn_tensor_load_to_lds(g0, g1, z4, z4, z8, 0);
#else
    __builtin_amdgcn_tensor_load_to_lds(g0, g1, z4, z4, 0);
#endif
}

// ---- fp32 -> bf16 conversion ----------------------------------------------
__global__ void cvt_f32_bf16(const float* __restrict__ in,
                             bf16_t* __restrict__ out, int n) {
    int i = blockIdx.x * blockDim.x + threadIdx.x;
    if (i < n) out[i] = (bf16_t)in[i];
}

// ---------------------------------------------------------------------------
// GEMM C[M,N] = A[M,K] * W[N,K]^T (+bias).  128x128 block, 8 waves (4Mx2N),
// wave tile 32x64.  K slabs of 32 staged by TDM, double buffered:
//   wave0 issues next-slab TDM -> all waves WMMA on current -> wave0
//   s_wait_tensorcnt 0 -> barrier -> flip.
// MODE 0: QKV epilogue -> Q(scaled)[b,h,n,d], K[b,h,n,d], V^T[b,h,d,n] (bf16)
// MODE 1: proj epilogue -> fp32 out + bias
// ---------------------------------------------------------------------------
template <int MODE>
__global__ __launch_bounds__(256)
void gemm_bf16_kernel(const bf16_t* __restrict__ A,
                      const bf16_t* __restrict__ W,
                      const float*  __restrict__ bias,
                      int M, int N, int K,
                      bf16_t* __restrict__ Qb, bf16_t* __restrict__ Kb,
                      bf16_t* __restrict__ Vt,
                      float*  __restrict__ Out) {
    __shared__ bf16_t As[2][128 * LSTR];
    __shared__ bf16_t Bs[2][128 * LSTR];

    const int tid  = threadIdx.x;
    const int wave = tid >> 5;
    const int lane = tid & 31;
    const int lo   = lane & 15;
    const int hi   = lane >> 4;

    const int bm = blockIdx.x * 128;
    const int bn = blockIdx.y * 128;
    const int wm = (wave & 3) * 32;
    const int wn = (wave >> 2) * 64;

    const bf16_t* Abase = A + (size_t)bm * K;
    const bf16_t* Wbase = W + (size_t)bn * K;

    v8f acc[2][4];
#pragma unroll
    for (int i = 0; i < 2; ++i)
#pragma unroll
        for (int j = 0; j < 4; ++j)
#pragma unroll
            for (int r = 0; r < 8; ++r) acc[i][j][r] = 0.0f;

    // prologue: TDM slab 0 into buffer 0
    if (wave == 0) {
        tdm_load_tile(lds_off(&As[0][0]), Abase, (uint32_t)K, (uint32_t)K);
        tdm_load_tile(lds_off(&Bs[0][0]), Wbase, (uint32_t)K, (uint32_t)K);
        __builtin_amdgcn_s_wait_tensorcnt((short)0);
    }
    __syncthreads();

    for (int kt = 0; kt < K; kt += 32) {
        const int cur = (kt >> 5) & 1;
        const bool more = (kt + 32) < K;
        if (wave == 0 && more) {   // async: overlaps the WMMAs below
            tdm_load_tile(lds_off(&As[1 - cur][0]), Abase + kt + 32,
                          (uint32_t)K, (uint32_t)K);
            tdm_load_tile(lds_off(&Bs[1 - cur][0]), Wbase + kt + 32,
                          (uint32_t)K, (uint32_t)K);
        }

        v16bf afr[2], bfr[4];
#pragma unroll
        for (int i = 0; i < 2; ++i)
            afr[i] = load_frag_a(&As[cur][0], LSTR, wm + i * 16 + lo, hi);
#pragma unroll
        for (int j = 0; j < 4; ++j)
            bfr[j] = load_frag_b(&Bs[cur][0], LSTR, wn + j * 16 + lo, hi);
#pragma unroll
        for (int i = 0; i < 2; ++i)
#pragma unroll
            for (int j = 0; j < 4; ++j)
                acc[i][j] = wmma_bf16(afr[i], bfr[j], acc[i][j]);

        if (wave == 0 && more)
            __builtin_amdgcn_s_wait_tensorcnt((short)0);
        __syncthreads();   // emits s_wait_dscnt before barrier -> safe flip
    }

    // Epilogue. C layout: lane L, vgpr r -> row (r + 8*hi), col (L%16).
#pragma unroll
    for (int j = 0; j < 4; ++j) {
        const int n   = bn + wn + j * 16 + lo;
        const float bv = bias[n];
#pragma unroll
        for (int i = 0; i < 2; ++i) {
#pragma unroll
            for (int r = 0; r < 8; ++r) {
                const int m   = bm + wm + i * 16 + r + 8 * hi;
                const float v = acc[i][j][r] + bv;
                if (MODE == 0) {
                    const int comp = n / DIM;
                    const int f    = n % DIM;
                    const int head = f >> 6;
                    const int d    = f & 63;
                    const int b    = m / NTOK;
                    const int q    = m % NTOK;
                    const size_t bh = (size_t)b * NH + head;
                    if (comp == 0)
                        Qb[(bh * NTOK + q) * HD + d] = (bf16_t)(v * SCALE);
                    else if (comp == 1)
                        Kb[(bh * NTOK + q) * HD + d] = (bf16_t)v;
                    else
                        Vt[(bh * HD + d) * NTOK + q] = (bf16_t)v;
                } else {
                    Out[(size_t)m * N + n] = v;
                }
            }
        }
    }
}

// ---------------------------------------------------------------------------
// Flash attention.  Wave = 16 q rows x HD=64.  Per 32-key block:
//   4 WMMAs (S) -> mask -> S(f32) to per-wave LDS -> reread in A layout
//   (lane = row) so softmax stats are scalar-per-lane with ONE shfl_xor(16)
//   each; P packs straight into the A fragment -> 4 WMMAs (O += P V).
//   K fragments software-pipelined one block ahead; V loads hoisted before
//   the softmax so global latency hides under exp/reduction VALU.
// ---------------------------------------------------------------------------
__global__ __launch_bounds__(256)
void attn_kernel(const bf16_t* __restrict__ Q,
                 const bf16_t* __restrict__ Kb,
                 const bf16_t* __restrict__ Vt,
                 const int*    __restrict__ mask,
                 bf16_t* __restrict__ Omid) {
    __shared__ float S_lds[8][16 * SSTR];   // per-wave private staging

    const int tid  = threadIdx.x;
    const int wave = tid >> 5;
    const int lane = tid & 31;
    const int lo   = lane & 15;
    const int hi   = lane >> 4;

    const int bh = blockIdx.y;
    const int b  = bh / NH;
    const int h  = bh % NH;
    const int q0 = blockIdx.x * 128 + wave * 16;

    const bf16_t* Qp = Q  + (size_t)bh * NTOK * HD;
    const bf16_t* Kp = Kb + (size_t)bh * NTOK * HD;
    const bf16_t* Vp = Vt + (size_t)bh * HD * NTOK;
    const int*    Mp = mask + (size_t)b * NTOK;

    v16bf aq[2];
#pragma unroll
    for (int kc = 0; kc < 2; ++kc)
        aq[kc] = load_frag_a(Qp + kc * 32, HD, q0 + lo, hi);

    v8f o[4];
#pragma unroll
    for (int t = 0; t < 4; ++t)
#pragma unroll
        for (int r = 0; r < 8; ++r) o[t][r] = 0.0f;
    float mi = -1e30f, li = 0.0f;      // scalar stats (lane = row)

    // pipeline: K fragments for block 0
    v16bf bk[2][2];
#pragma unroll
    for (int kc = 0; kc < 2; ++kc)
#pragma unroll
        for (int t = 0; t < 2; ++t)
            bk[kc][t] = load_frag_b(Kp + kc * 32, HD, t * 16 + lo, hi);

    float* Sw = &S_lds[wave][0];

    for (int kb = 0; kb < NTOK; kb += 32) {
        // ---- S = Q K^T (current block, prefetched fragments)
        v8f s0, s1;
#pragma unroll
        for (int r = 0; r < 8; ++r) { s0[r] = 0.0f; s1[r] = 0.0f; }
#pragma unroll
        for (int kc = 0; kc < 2; ++kc) {
            s0 = wmma_bf16(aq[kc], bk[kc][0], s0);
            s1 = wmma_bf16(aq[kc], bk[kc][1], s1);
        }

        // ---- issue next block's K fragments + this block's V fragments now;
        //      they complete under the softmax VALU work below.
        const int kn = (kb + 32 < NTOK) ? kb + 32 : kb;
        v16bf nbk[2][2];
#pragma unroll
        for (int kc = 0; kc < 2; ++kc)
#pragma unroll
            for (int t = 0; t < 2; ++t)
                nbk[kc][t] = load_frag_b(Kp + kc * 32, HD, kn + t * 16 + lo, hi);
        v16bf bv[4];
#pragma unroll
        for (int t = 0; t < 4; ++t)
            bv[t] = load_frag_b(Vp + kb, NTOK, t * 16 + lo, hi);

        // ---- mask (C layout: lane's column = key), branch-free
        const int mk0 = Mp[kb + lo];
        const int mk1 = Mp[kb + 16 + lo];
#pragma unroll
        for (int r = 0; r < 8; ++r) {
            s0[r] = mk0 ? s0[r] : -1e30f;
            s1[r] = mk1 ? s1[r] : -1e30f;
        }

        // ---- stage S to LDS in C layout positions
#pragma unroll
        for (int r = 0; r < 8; ++r) {
            Sw[(r + 8 * hi) * SSTR + lo]      = s0[r];
            Sw[(r + 8 * hi) * SSTR + 16 + lo] = s1[r];
        }
        asm volatile("s_wait_dscnt 0x0" ::: "memory");
        __builtin_amdgcn_wave_barrier();

        // ---- reread row `lo` in A-fragment element order (16B-aligned b128s)
        union { float f[16]; float4 q4[4]; } sv;
        const float* Srow = Sw + lo * SSTR + hi * 8;
        sv.q4[0] = *(const float4*)(Srow);
        sv.q4[1] = *(const float4*)(Srow + 4);
        sv.q4[2] = *(const float4*)(Srow + 16);
        sv.q4[3] = *(const float4*)(Srow + 20);

        // ---- scalar online softmax: one shfl each for max and sum
        float rm = sv.f[0];
#pragma unroll
        for (int e = 1; e < 16; ++e) rm = fmaxf(rm, sv.f[e]);
        rm = fmaxf(rm, __shfl_xor(rm, 16, 32));
        const float mn    = fmaxf(mi, rm);
        const float alpha = __expf(mi - mn);
        mi = mn;
        float rs = 0.0f;
        v16bf ap;
#pragma unroll
        for (int e = 0; e < 16; ++e) {
            const float p = __expf(sv.f[e] - mn);
            rs += p;
            ap[e] = (bf16_t)p;      // lands directly in A-fragment order
        }
        rs += __shfl_xor(rs, 16, 32);
        li = li * alpha + rs;

        // ---- broadcast alpha (lives in lane=row) to C-layout rows r+8*hi
        float al[8];
#pragma unroll
        for (int r = 0; r < 8; ++r) al[r] = __shfl(alpha, 8 * hi + r, 32);
#pragma unroll
        for (int t = 0; t < 4; ++t)
#pragma unroll
            for (int r = 0; r < 8; ++r) o[t][r] *= al[r];

        // ---- O += P V
#pragma unroll
        for (int t = 0; t < 4; ++t)
            o[t] = wmma_bf16(ap, bv[t], o[t]);

        // rotate K pipeline
#pragma unroll
        for (int kc = 0; kc < 2; ++kc)
#pragma unroll
            for (int t = 0; t < 2; ++t) bk[kc][t] = nbk[kc][t];
    }

    // ---- normalize (broadcast 1/li to C-layout rows) and store bf16
    const float inv = 1.0f / li;
    float ir[8];
#pragma unroll
    for (int r = 0; r < 8; ++r) ir[r] = __shfl(inv, 8 * hi + r, 32);
#pragma unroll
    for (int r = 0; r < 8; ++r) {
        const int q = q0 + r + 8 * hi;
#pragma unroll
        for (int t = 0; t < 4; ++t) {
            const int d = t * 16 + lo;
            Omid[((size_t)b * NTOK + q) * DIM + h * HD + d] =
                (bf16_t)(o[t][r] * ir[r]);
        }
    }
}

// ---------------------------------------------------------------------------
extern "C" void kernel_launch(void* const* d_in, const int* in_sizes, int n_in,
                              void* d_out, int out_size, void* d_ws, size_t ws_size,
                              hipStream_t stream) {
    const float* x      = (const float*)d_in[0];
    const int*   mask   = (const int*)  d_in[1];
    const float* qkv_w  = (const float*)d_in[2];
    const float* qkv_b  = (const float*)d_in[3];
    const float* proj_w = (const float*)d_in[4];
    const float* proj_b = (const float*)d_in[5];
    float* out = (float*)d_out;

    const int M = BATCH * NTOK;                 // 8192
    const size_t nX  = (size_t)M * DIM;
    const size_t nWq = (size_t)QKVN * DIM;
    const size_t nWp = (size_t)DIM * DIM;
    const size_t nHd = (size_t)BATCH * NH * NTOK * HD;

    char* ws = (char*)d_ws;
    size_t off = 0;
    auto take = [&](size_t elems) {
        char* p = ws + off;
        off = (off + elems * sizeof(bf16_t) + 255) & ~(size_t)255;
        return (bf16_t*)p;
    };
    bf16_t* xs  = take(nX);
    bf16_t* wq  = take(nWq);
    bf16_t* wp  = take(nWp);
    bf16_t* Qb  = take(nHd);
    bf16_t* Kbf = take(nHd);
    bf16_t* Vt  = take(nHd);
    bf16_t* mid = take(nX);

    cvt_f32_bf16<<<(int)((nX  + 255) / 256), 256, 0, stream>>>(x,      xs, (int)nX);
    cvt_f32_bf16<<<(int)((nWq + 255) / 256), 256, 0, stream>>>(qkv_w,  wq, (int)nWq);
    cvt_f32_bf16<<<(int)((nWp + 255) / 256), 256, 0, stream>>>(proj_w, wp, (int)nWp);

    gemm_bf16_kernel<0><<<dim3(M / 128, QKVN / 128), 256, 0, stream>>>(
        xs, wq, qkv_b, M, QKVN, DIM, Qb, Kbf, Vt, nullptr);

    attn_kernel<<<dim3(NTOK / 128, BATCH * NH), 256, 0, stream>>>(
        Qb, Kbf, Vt, mask, mid);

    gemm_bf16_kernel<1><<<dim3(M / 128, DIM / 128), 256, 0, stream>>>(
        mid, wp, proj_b, M, DIM, DIM, nullptr, nullptr, nullptr, out);
}
